// HyperAttention_6004364280630
// MI455X (gfx1250) — compile-verified, matrix operand order checked by reading, not run on hardware
//
#include <hip/hip_runtime.h>
#include <hip/hip_bf16.h>
#include <cstdint>
#include <math.h>

// ---------------------------------------------------------------------------
// Sedenion window-GEMM + GELU + projection for MI455X (gfx1250, wave32, WMMA)
//
// GEMM1: xw(32768x1536) @ big(1536x1536), f16 in / f32 accum via
//        v_wmma_f32_16x16x32_f16, A/B staged with global_load_async_to_lds_b128
//        (ASYNCcnt) into double-buffered LDS. GEMM2: z(524288x96) @ pw^T.
// Workspace (d_ws) layout, ~197 MB total:
//   [0)            xw    f16  32768*1536   (windowed activations)
//   [100663296)    bigT  f16  1536*1536    (sedenion matrix, TRANSPOSED [n][k])
//   [105381888)    pw    f16  96*96        (proj_w row-major == B[n][k])
//   [105400320)    z     f16  524288*96    (GELU'd pixel-major intermediate)
// ---------------------------------------------------------------------------

typedef _Float16 v16h __attribute__((ext_vector_type(16)));
typedef _Float16 v8h  __attribute__((ext_vector_type(8)));
typedef float    v8f  __attribute__((ext_vector_type(8)));

union V16 { v16h v; v8h h[2]; };

#define CCH 96
#define K1  1536

// --- CDNA5 async global->LDS copy (16B per lane, tracked by ASYNCcnt) ------
__device__ __forceinline__ void async_ld16(const void* g, void* lds) {
  unsigned lo = (unsigned)(uintptr_t)lds;          // generic LDS ptr: low 32b = LDS addr
  unsigned long long ga = (unsigned long long)(uintptr_t)g;
  asm volatile("global_load_async_to_lds_b128 %0, %1, off"
               :: "v"(lo), "v"(ga) : "memory");
}
__device__ __forceinline__ void wait_async0() {
  asm volatile("s_wait_asynccnt 0x0" ::: "memory");
}

// Iterative Cayley-Dickson table: e_i * e_j = sign * e_idx  (n=16)
__device__ __forceinline__ void cd16(int i, int j, int& sign, int& idx) {
  int s = 1, kofs = 0, n = 16;
  while (n > 1) {
    int m = n >> 1;
    if (i < m && j < m) {
    } else if (i < m) {            // (a,0)(0,d) -> recurse (q, i), +m
      int q = j - m; j = i; i = q; kofs += m;
    } else if (j < m) {            // (0,b)(c,0) -> recurse (p, j)*conj(j), +m
      i -= m; if (j != 0) s = -s; kofs += m;
    } else {                       // (0,b)(0,d) -> -conj(q) * recurse(q, p)
      int p = i - m, q = j - m; if (q == 0) s = -s; i = q; j = p;
    }
    n = m;
  }
  sign = s; idx = kofs;
}

// bigT[(k*96+cc)*1536 + (j*96+r)] = SIGN[i,j]*w_comp[i][r][cc]  (N-major!)
__global__ __launch_bounds__(256) void build_big(const float* __restrict__ wc,
                                                 _Float16* __restrict__ bigT) {
  int idx = blockIdx.x * 256 + threadIdx.x;          // idx = col*1536 + row
  if (idx >= K1 * K1) return;
  int col = idx / K1, row = idx % K1;                // row = K index, col = N index
  int j = row / CCH, r = row % CCH;
  int k = col / CCH, cc = col % CCH;
  float v = 0.f;
  #pragma unroll 1
  for (int i = 0; i < 16; ++i) {
    int s, kk; cd16(i, j, s, kk);
    if (kk == k) { v = (float)s * wc[(i * CCH + r) * CCH + cc]; break; }
  }
  bigT[idx] = (_Float16)v;
}

__global__ __launch_bounds__(256) void conv_pw(const float* __restrict__ pwf,
                                               _Float16* __restrict__ pw) {
  int idx = blockIdx.x * 256 + threadIdx.x;
  if (idx < CCH * CCH) pw[idx] = (_Float16)pwf[idx];
}

// One thread per (b,c,y,ww): coalesced float4 read of x, 4 f16 scatter-stores.
// xw[m][(s1*4+s2)*96+c] = x[b][c][4h+s1][4w+s2]
__global__ __launch_bounds__(256) void window_f16(const float* __restrict__ x,
                                                  _Float16* __restrict__ xw) {
  int idx = blockIdx.x * 256 + threadIdx.x;          // 12582912 total
  int ww = idx & 63;
  int y  = (idx >> 6) & 255;
  int bc = idx >> 14;                                // b*96 + c
  int c = bc % CCH, b = bc / CCH;
  float4 v = *(const float4*)&x[(size_t)idx * 4];    // x is contiguous here
  int hh = y >> 2, s1 = y & 3;
  int m = b * 4096 + hh * 64 + ww;
  size_t base = (size_t)m * K1 + s1 * 384 + c;
  xw[base        ] = (_Float16)v.x;
  xw[base +  96  ] = (_Float16)v.y;
  xw[base + 192  ] = (_Float16)v.z;
  xw[base + 288  ] = (_Float16)v.w;
}

__device__ __forceinline__ float gelu_exact(float v) {
  return 0.5f * v * (1.0f + erff(v * 0.70710678118654752f));
}

// GEMM1: 64(M) x 96(N) tile per block, 8 waves of 16x48, K-step 32.
// Double-buffered async LDS staging; epilogue: GELU -> f16 pixel-major z.
__global__ __launch_bounds__(256) void gemm1(const _Float16* __restrict__ xw,
                                             const _Float16* __restrict__ bigT,
                                             _Float16* __restrict__ z) {
  __shared__ __align__(16) _Float16 smA[2][64 * 32];
  __shared__ __align__(16) _Float16 smB[2][96 * 32];
  __shared__ __align__(16) _Float16 zs[64 * 96];

  const int tid  = threadIdx.x;
  const int lane = tid & 31, wave = tid >> 5;
  const int wm = wave & 3, wn = wave >> 2;            // 4x2 wave grid
  const int lrow = lane & 15, hig = lane >> 4, kb0 = hig * 8;
  const int m0   = blockIdx.x * 64;
  const int sblk = blockIdx.y;
  const int n0   = sblk * 96;

  const int arow_st = tid >> 2, aseg = tid & 3;       // A stage: 256 x 16B
  const int brow_st = tid >> 2, bseg = tid & 3;       // B stage: 384 x 16B

  auto issue = [&](int kb, int buf) {
    async_ld16(&xw[(size_t)(m0 + arow_st) * K1 + kb + aseg * 8],
               &smA[buf][arow_st * 32 + aseg * 8]);
    async_ld16(&bigT[(size_t)(n0 + brow_st) * K1 + kb + bseg * 8],
               &smB[buf][brow_st * 32 + bseg * 8]);
    if (tid < 128) {
      int id2 = tid + 256, n2 = id2 >> 2, sg2 = id2 & 3;
      async_ld16(&bigT[(size_t)(n0 + n2) * K1 + kb + sg2 * 8],
                 &smB[buf][n2 * 32 + sg2 * 8]);
    }
  };

  v8f acc[3] = {};
  issue(0, 0);
  int buf = 0;
  for (int kb = 0; kb < K1; kb += 32) {
    wait_async0();                 // my async writes to LDS are done
    __syncthreads();               // everyone's are visible
    if (kb + 32 < K1) {
      issue(kb + 32, buf ^ 1);     // overlap next tile's DMA with compute
      __builtin_prefetch(&xw[(size_t)(m0 + arow_st) * K1 + kb + 64 + aseg * 8], 0, 1);
    }
    V16 a;
    int arow = wm * 16 + lrow;
    a.h[0] = *(const v8h*)&smA[buf][arow * 32 + kb0];
    a.h[1] = *(const v8h*)&smA[buf][arow * 32 + 16 + kb0];
    #pragma unroll
    for (int t = 0; t < 3; ++t) {
      int bn = wn * 48 + t * 16 + lrow;
      V16 bf;
      bf.h[0] = *(const v8h*)&smB[buf][bn * 32 + kb0];
      bf.h[1] = *(const v8h*)&smB[buf][bn * 32 + 16 + kb0];
      acc[t] = __builtin_amdgcn_wmma_f32_16x16x32_f16(false, a.v, false, bf.v,
                                                      (short)0, acc[t], false, false);
    }
    buf ^= 1;
  }

  // ---- epilogue: GELU, LDS restage, pixel-major coalesced dword stores ----
  __syncthreads();
  #pragma unroll
  for (int t = 0; t < 3; ++t) {
    int col = wn * 48 + t * 16 + lrow;
    #pragma unroll
    for (int r = 0; r < 8; ++r) {
      int mloc = wm * 16 + hig * 8 + r;               // C/D layout: M = r + hig*8
      zs[mloc * 96 + col] = (_Float16)gelu_exact(acc[t][r]);
    }
  }
  __syncthreads();
  const int s1 = sblk >> 2, s2 = sblk & 3;
  const uint32_t* zs32 = (const uint32_t*)zs;
  uint32_t* z32 = (uint32_t*)z;
  #pragma unroll
  for (int it = 0; it < 12; ++it) {
    int id2 = tid + it * 256;
    int mloc = id2 / 48, d = id2 % 48;                // 48 dwords = one pixel row
    int m = m0 + mloc;
    int b = m >> 12, hh = (m >> 6) & 63, ww = m & 63;
    int pix = b * 65536 + (hh * 4 + s1) * 256 + ww * 4 + s2;
    z32[(size_t)pix * 48 + d] = zs32[mloc * 48 + d];
  }
}

// GEMM2: 64 pixels x 96 channels; K=96 in 3 WMMA k-steps. B (proj_w f16,
// 18KB, cache-resident) fragments come straight from global; A staged async.
__global__ __launch_bounds__(256) void gemm2(const _Float16* __restrict__ z,
                                             const _Float16* __restrict__ pw,
                                             const float* __restrict__ bias,
                                             float* __restrict__ out) {
  __shared__ __align__(16) _Float16 smA[64 * 32];
  __shared__ float smO[64 * 97];                      // padded rows

  const int tid  = threadIdx.x;
  const int lane = tid & 31, wave = tid >> 5;
  const int wm = wave & 3, wn = wave >> 2;
  const int lrow = lane & 15, hig = lane >> 4, kb0 = hig * 8;
  const int m0 = blockIdx.x * 64;                     // pixel base

  v8f acc[3] = {};
  #pragma unroll
  for (int kb = 0; kb < 96; kb += 32) {
    __syncthreads();
    { int row = tid >> 2, seg = tid & 3;
      async_ld16(&z[(size_t)(m0 + row) * 96 + kb + seg * 8],
                 &smA[row * 32 + seg * 8]); }
    wait_async0();
    __syncthreads();

    V16 a;
    int arow = wm * 16 + lrow;
    a.h[0] = *(const v8h*)&smA[arow * 32 + kb0];
    a.h[1] = *(const v8h*)&smA[arow * 32 + 16 + kb0];
    #pragma unroll
    for (int t = 0; t < 3; ++t) {
      int bn = wn * 48 + t * 16 + lrow;
      V16 bf;                                         // pw[n][k]: fragment-ready
      bf.h[0] = *(const v8h*)&pw[bn * 96 + kb + kb0];
      bf.h[1] = *(const v8h*)&pw[bn * 96 + kb + 16 + kb0];
      acc[t] = __builtin_amdgcn_wmma_f32_16x16x32_f16(false, a.v, false, bf.v,
                                                      (short)0, acc[t], false, false);
    }
  }

  __syncthreads();
  #pragma unroll
  for (int t = 0; t < 3; ++t) {
    int col = wn * 48 + t * 16 + lrow;
    float bv = bias[col];
    #pragma unroll
    for (int r = 0; r < 8; ++r) {
      int mloc = wm * 16 + hig * 8 + r;
      smO[mloc * 97 + col] = acc[t][r] + bv;
    }
  }
  __syncthreads();
  // out[b][c][y][x]: 64 consecutive x per (b,c) -> 256B contiguous stores
  #pragma unroll
  for (int it = 0; it < 24; ++it) {
    int id2 = tid + it * 256;
    int c = id2 >> 6, xx = id2 & 63;
    int p = m0 + xx;                                  // pixel = b*65536 + y*256 + x
    int b = p >> 16;
    out[((size_t)(b * 96 + c) << 16) + (p & 65535)] = smO[xx * 97 + c];
  }
}

extern "C" void kernel_launch(void* const* d_in, const int* in_sizes, int n_in,
                              void* d_out, int out_size, void* d_ws, size_t ws_size,
                              hipStream_t stream) {
  const float* x      = (const float*)d_in[0];   // (8,96,256,256)
  const float* w_comp = (const float*)d_in[1];   // (16,96,96)
  const float* proj_w = (const float*)d_in[2];   // (96,96)
  const float* proj_b = (const float*)d_in[3];   // (96,)
  float* out = (float*)d_out;

  char* ws = (char*)d_ws;
  _Float16* xw   = (_Float16*)(ws);                    // 100663296 B
  _Float16* bigT = (_Float16*)(ws + 100663296);        //   4718592 B
  _Float16* pw   = (_Float16*)(ws + 105381888);        //     18432 B
  _Float16* z    = (_Float16*)(ws + 105400320);        // 100663296 B

  build_big <<<9216,  256, 0, stream>>>(w_comp, bigT);
  conv_pw   <<<36,    256, 0, stream>>>(proj_w, pw);
  window_f16<<<49152, 256, 0, stream>>>(x, xw);
  gemm1     <<<dim3(512, 16), 256, 0, stream>>>(xw, bigT, z);
  gemm2     <<<8192,  256, 0, stream>>>(z, pw, proj_b, out);
}
